// BiasedEncoderLayer_3805341024354
// MI455X (gfx1250) — matrix operand order repaired and checked by reading, not compiled, not run on hardware
//
#include <hip/hip_runtime.h>
#include <hip/hip_bf16.h>

// ---------------------------------------------------------------------------
// BiasedEncoderLayer for MI455X (gfx1250), wave32 + v_wmma_f32_16x16x32_f16.
// D=256, H=8, HD=32, DFF=1024, S=1024, B=8. f32 I/O, f16 WMMA operands,
// f32 accumulation. GEMMs register-blocked 4x2 tiles (64x32 out) per wave.
// ---------------------------------------------------------------------------

#define S_LEN 1024
#define BATCH 8
#define DMODEL 256
#define NHEAD 8
#define HDIM 32
#define DFF 1024
#define ROWS (S_LEN * BATCH)   // 8192
#define EPSV 1e-5f
#define QSCALE 0.17677669529663687f  // 32^-0.5

typedef _Float16 v16h __attribute__((ext_vector_type(16)));
typedef _Float16 v8h  __attribute__((ext_vector_type(8)));
typedef float    v8f  __attribute__((ext_vector_type(8)));

union V16 { v16h v; v8h h[2]; };

// ---------------- wave-level reductions (wave32) ----------------
__device__ __forceinline__ float redmax16(float v) {
    v = fmaxf(v, __shfl_xor(v, 1, 32));
    v = fmaxf(v, __shfl_xor(v, 2, 32));
    v = fmaxf(v, __shfl_xor(v, 4, 32));
    v = fmaxf(v, __shfl_xor(v, 8, 32));
    return v;
}
__device__ __forceinline__ float redsum16(float v) {
    v += __shfl_xor(v, 1, 32);
    v += __shfl_xor(v, 2, 32);
    v += __shfl_xor(v, 4, 32);
    v += __shfl_xor(v, 8, 32);
    return v;
}
__device__ __forceinline__ float redsum32(float v) {
    v = redsum16(v);
    v += __shfl_xor(v, 16, 32);
    return v;
}

// ---------------- f32 -> f16 conversion ----------------
__global__ void cvt_f16_kernel(const float* __restrict__ src,
                               _Float16* __restrict__ dst, int n) {
    int i = blockIdx.x * blockDim.x + threadIdx.x;
    if (i < n) dst[i] = (_Float16)src[i];
}

// ---------------- LayerNorm over D=256 (one wave per row) ----------------
__global__ void ln_kernel(const float* __restrict__ in,
                          const float* __restrict__ g,
                          const float* __restrict__ be,
                          float* __restrict__ outf,
                          _Float16* __restrict__ outh) {
    int row  = blockIdx.x * 8 + (threadIdx.x >> 5);
    int lane = threadIdx.x & 31;
    const float* p = in + (size_t)row * DMODEL + lane * 8;
    float4 v0 = *(const float4*)(p);
    float4 v1 = *(const float4*)(p + 4);
    float x[8] = {v0.x, v0.y, v0.z, v0.w, v1.x, v1.y, v1.z, v1.w};
    float s = 0.f, sq = 0.f;
#pragma unroll
    for (int i = 0; i < 8; ++i) { s += x[i]; sq += x[i] * x[i]; }
    s  = redsum32(s);
    sq = redsum32(sq);
    float mean = s * (1.0f / DMODEL);
    float var  = sq * (1.0f / DMODEL) - mean * mean;
    float rstd = rsqrtf(var + EPSV);
#pragma unroll
    for (int i = 0; i < 8; ++i) {
        int c = lane * 8 + i;
        float o = (x[i] - mean) * rstd * g[c] + be[c];
        outf[(size_t)row * DMODEL + c] = o;
        outh[(size_t)row * DMODEL + c] = (_Float16)o;
    }
}

// ---------------- blocked WMMA GEMM: C[M,N] = A[M,K] * W[N,K]^T + bias ------
// Each wave computes a 64x32 output block (4 M-tiles x 2 N-tiles):
// per K-step: 4 A frags + 2 B frags -> 8 independent WMMAs.
// EP 0: QKV scatter-to-heads (q scaled, V stored transposed), f16 out
// EP 1: f32 out = acc + bias + resid  (residual add)
// EP 2: f16 out = relu(acc + bias)
template <int EP>
__global__ void gemm_kernel(const _Float16* __restrict__ A,
                            const _Float16* __restrict__ W,
                            const float* __restrict__ bias,
                            int M, int N, int K,
                            float* __restrict__ outf,
                            _Float16* __restrict__ outh,
                            const float* __restrict__ resid,
                            _Float16* __restrict__ qh,
                            _Float16* __restrict__ kh,
                            _Float16* __restrict__ vt) {
    constexpr int MB = 4, NB = 2;
    int lane = threadIdx.x & 31;
    int wave = threadIdx.x >> 5;
    int bn = N / (16 * NB);
    int blk = blockIdx.x * 8 + wave;
    int tmB = blk / bn;           // 64-row block
    int tnB = blk % bn;           // 32-col block
    int m16 = lane & 15;
    int khf = lane >> 4;

    const _Float16* abase = A + (size_t)(tmB * 64 + m16) * K + khf * 8;
    const _Float16* wbase = W + (size_t)(tnB * 32 + m16) * K + khf * 8;

    v8f acc[MB][NB];
#pragma unroll
    for (int i = 0; i < MB; ++i)
#pragma unroll
        for (int j = 0; j < NB; ++j) acc[i][j] = (v8f){};

    for (int k0 = 0; k0 < K; k0 += 32) {
        V16 a[MB], b[NB];
#pragma unroll
        for (int i = 0; i < MB; ++i) {
            const _Float16* p = abase + (size_t)(i * 16) * K + k0;
            a[i].h[0] = *(const v8h*)(p);
            a[i].h[1] = *(const v8h*)(p + 16);
        }
#pragma unroll
        for (int j = 0; j < NB; ++j) {
            const _Float16* p = wbase + (size_t)(j * 16) * K + k0;
            b[j].h[0] = *(const v8h*)(p);
            b[j].h[1] = *(const v8h*)(p + 16);
        }
#pragma unroll
        for (int i = 0; i < MB; ++i)
#pragma unroll
            for (int j = 0; j < NB; ++j)
                acc[i][j] = __builtin_amdgcn_wmma_f32_16x16x32_f16(
                    false, a[i].v, false, b[j].v, (short)0, acc[i][j],
                    false, false);
    }

#pragma unroll
    for (int j = 0; j < NB; ++j) {
        int nn = tnB * 32 + j * 16 + m16;
        float bnv = bias[nn];
#pragma unroll
        for (int i = 0; i < MB; ++i) {
#pragma unroll
            for (int r = 0; r < 8; ++r) {
                int mm = tmB * 64 + i * 16 + r + 8 * khf;
                float val = acc[i][j][r] + bnv;
                if (EP == 0) {
                    int s = mm >> 3;   // row = s*B + b, B=8
                    int bb = mm & 7;
                    if (nn < DMODEL) {
                        int h = nn >> 5, hd = nn & 31;
                        qh[(((size_t)(bb * NHEAD + h)) * S_LEN + s) * HDIM + hd] =
                            (_Float16)(val * QSCALE);
                    } else if (nn < 2 * DMODEL) {
                        int c = nn - DMODEL, h = c >> 5, hd = c & 31;
                        kh[(((size_t)(bb * NHEAD + h)) * S_LEN + s) * HDIM + hd] =
                            (_Float16)val;
                    } else {
                        int c = nn - 2 * DMODEL, h = c >> 5, hd = c & 31;
                        vt[(((size_t)(bb * NHEAD + h)) * HDIM + hd) * S_LEN + s] =
                            (_Float16)val;
                    }
                } else if (EP == 1) {
                    outf[(size_t)mm * N + nn] = val + resid[(size_t)mm * N + nn];
                } else {  // EP == 2
                    outh[(size_t)mm * N + nn] = (_Float16)fmaxf(val, 0.f);
                }
            }
        }
    }
}

// ---------------- fused flash attention (one wave = 16 query rows) ----------
__global__ void attn_kernel(const _Float16* __restrict__ Qh,
                            const _Float16* __restrict__ Kh,
                            const _Float16* __restrict__ Vt,
                            const float* __restrict__ bias,
                            _Float16* __restrict__ ctx_h) {
    __shared__ __align__(16) _Float16 pbuf[8 * 16 * 32];  // 1KB per wave
    int lane = threadIdx.x & 31;
    int wave = threadIdx.x >> 5;
    int gw = blockIdx.x * 8 + wave;
    int st = gw & 63;       // 64 query tiles per (b,h)
    int bh = gw >> 6;       // 0..63
    int b = bh >> 3, h = bh & 7;
    int m16 = lane & 15;
    int khf = lane >> 4;

    // Q fragment (A-layout, K=HD=32), loaded once
    const _Float16* qbase = Qh + (size_t)bh * S_LEN * HDIM;
    int qrow = st * 16 + m16;
    V16 qA;
    qA.h[0] = *(const v8h*)(qbase + (size_t)qrow * HDIM + khf * 8);
    qA.h[1] = *(const v8h*)(qbase + (size_t)qrow * HDIM + khf * 8 + 16);

    const float* bptr = bias + (size_t)(b * NHEAD + h) * S_LEN * S_LEN;
    const _Float16* kb = Kh + (size_t)bh * S_LEN * HDIM;
    const _Float16* vb = Vt + (size_t)bh * HDIM * S_LEN;
    _Float16* pb = pbuf + wave * 512;

    v8f acc0 = {}, acc1 = {};
    float mrun[8], lrun[8];
#pragma unroll
    for (int r = 0; r < 8; ++r) { mrun[r] = -INFINITY; lrun[r] = 0.f; }

    for (int t0 = 0; t0 < S_LEN; t0 += 32) {
        // K fragments (B-layout): lane = key column, contiguous K reads
        V16 kB0, kB1;
        {
            const _Float16* p0 = kb + (size_t)(t0 + m16) * HDIM + khf * 8;
            const _Float16* p1 = kb + (size_t)(t0 + 16 + m16) * HDIM + khf * 8;
            kB0.h[0] = *(const v8h*)(p0); kB0.h[1] = *(const v8h*)(p0 + 16);
            kB1.h[0] = *(const v8h*)(p1); kB1.h[1] = *(const v8h*)(p1 + 16);
        }
        v8f z = {};
        v8f s0 = __builtin_amdgcn_wmma_f32_16x16x32_f16(
            false, qA.v, false, kB0.v, (short)0, z, false, false);
        v8f s1 = __builtin_amdgcn_wmma_f32_16x16x32_f16(
            false, qA.v, false, kB1.v, (short)0, z, false, false);

        // bias add + online softmax, P staged to LDS (wave-private tile)
#pragma unroll
        for (int r = 0; r < 8; ++r) {
            int m = st * 16 + r + 8 * khf;
            const float* brow = bptr + (size_t)m * S_LEN + t0;
            float v0 = s0[r] + brow[m16];
            float v1 = s1[r] + brow[16 + m16];
            if (t0 + 32 < S_LEN) __builtin_prefetch(brow + 32 + m16, 0, 0);
            float mx = redmax16(fmaxf(v0, v1));
            float mn = fmaxf(mrun[r], mx);
            float scale = __expf(mrun[r] - mn);
            float p0 = __expf(v0 - mn);
            float p1 = __expf(v1 - mn);
            float rs = redsum16(p0 + p1);
            lrun[r] = lrun[r] * scale + rs;
            mrun[r] = mn;
            acc0[r] *= scale;
            acc1[r] *= scale;
            int mrow = r + 8 * khf;
            pb[mrow * 32 + m16]      = (_Float16)p0;
            pb[mrow * 32 + 16 + m16] = (_Float16)p1;
        }
        // Intra-wave DS ordering only (tile is wave-private): LDS ops are
        // in-order per wave; wait for stores to land before re-reading.
        asm volatile("s_wait_dscnt 0x0" ::: "memory");
        // reload P in A-fragment layout
        V16 pA;
        pA.h[0] = *(const v8h*)(pb + m16 * 32 + khf * 8);
        pA.h[1] = *(const v8h*)(pb + m16 * 32 + 16 + khf * 8);

        // ctx += P(16x32) x V(32x16) for both HD column halves (V transposed)
        V16 vB0, vB1;
        {
            const _Float16* p0 = vb + (size_t)m16 * S_LEN + t0 + khf * 8;
            const _Float16* p1 = vb + (size_t)(16 + m16) * S_LEN + t0 + khf * 8;
            vB0.h[0] = *(const v8h*)(p0); vB0.h[1] = *(const v8h*)(p0 + 16);
            vB1.h[0] = *(const v8h*)(p1); vB1.h[1] = *(const v8h*)(p1 + 16);
        }
        acc0 = __builtin_amdgcn_wmma_f32_16x16x32_f16(
            false, pA.v, false, vB0.v, (short)0, acc0, false, false);
        acc1 = __builtin_amdgcn_wmma_f32_16x16x32_f16(
            false, pA.v, false, vB1.v, (short)0, acc1, false, false);
    }

    // normalize and scatter ctx -> [S,B,D] (f16 for out-proj GEMM)
#pragma unroll
    for (int r = 0; r < 8; ++r) {
        float inv = 1.0f / lrun[r];
        int m = st * 16 + r + 8 * khf;  // query index s
        size_t base = ((size_t)m * BATCH + b) * DMODEL + h * HDIM;
        ctx_h[base + m16]      = (_Float16)(acc0[r] * inv);
        ctx_h[base + 16 + m16] = (_Float16)(acc1[r] * inv);
    }
}

// ---------------------------------------------------------------------------
extern "C" void kernel_launch(void* const* d_in, const int* in_sizes, int n_in,
                              void* d_out, int out_size, void* d_ws, size_t ws_size,
                              hipStream_t stream) {
    const float* src       = (const float*)d_in[0];
    const float* bias      = (const float*)d_in[1];
    const float* in_proj_w = (const float*)d_in[2];
    const float* in_proj_b = (const float*)d_in[3];
    const float* out_w     = (const float*)d_in[4];
    const float* out_b     = (const float*)d_in[5];
    const float* w1        = (const float*)d_in[6];
    const float* b1        = (const float*)d_in[7];
    const float* w2        = (const float*)d_in[8];
    const float* b2        = (const float*)d_in[9];
    const float* g1        = (const float*)d_in[10];
    const float* be1       = (const float*)d_in[11];
    const float* g2        = (const float*)d_in[12];
    const float* be2       = (const float*)d_in[13];
    float* out = (float*)d_out;

    // ---- workspace layout (bytes) ----
    char* ws = (char*)d_ws;
    size_t off = 0;
    auto alloc = [&](size_t bytes) { char* p = ws + off; off += (bytes + 255) & ~(size_t)255; return p; };
    float*    x_f32  = (float*)alloc((size_t)ROWS * DMODEL * 4);
    float*    x2_f32 = (float*)alloc((size_t)ROWS * DMODEL * 4);
    float*    y_f32  = (float*)alloc((size_t)ROWS * DMODEL * 4);
    _Float16* x_h    = (_Float16*)alloc((size_t)ROWS * DMODEL * 2);
    _Float16* y_h    = (_Float16*)alloc((size_t)ROWS * DMODEL * 2);
    _Float16* qh     = (_Float16*)alloc((size_t)BATCH * NHEAD * S_LEN * HDIM * 2);
    _Float16* kh     = (_Float16*)alloc((size_t)BATCH * NHEAD * S_LEN * HDIM * 2);
    _Float16* vt     = (_Float16*)alloc((size_t)BATCH * NHEAD * S_LEN * HDIM * 2);
    _Float16* ctx_h  = (_Float16*)alloc((size_t)ROWS * DMODEL * 2);
    _Float16* h1_h   = (_Float16*)alloc((size_t)ROWS * DFF * 2);
    _Float16* win_h  = (_Float16*)alloc((size_t)3 * DMODEL * DMODEL * 2);
    _Float16* wout_h = (_Float16*)alloc((size_t)DMODEL * DMODEL * 2);
    _Float16* w1_h   = (_Float16*)alloc((size_t)DFF * DMODEL * 2);
    _Float16* w2_h   = (_Float16*)alloc((size_t)DMODEL * DFF * 2);
    (void)ws_size; (void)in_sizes; (void)n_in; (void)out_size;

    // 1) weight conversions to f16
    int nw;
    nw = 3 * DMODEL * DMODEL;
    cvt_f16_kernel<<<(nw + 255) / 256, 256, 0, stream>>>(in_proj_w, win_h, nw);
    nw = DMODEL * DMODEL;
    cvt_f16_kernel<<<(nw + 255) / 256, 256, 0, stream>>>(out_w, wout_h, nw);
    nw = DFF * DMODEL;
    cvt_f16_kernel<<<(nw + 255) / 256, 256, 0, stream>>>(w1, w1_h, nw);
    nw = DMODEL * DFF;
    cvt_f16_kernel<<<(nw + 255) / 256, 256, 0, stream>>>(w2, w2_h, nw);

    // 2) LN1: x = LN(src)
    ln_kernel<<<ROWS / 8, 256, 0, stream>>>(src, g1, be1, x_f32, x_h);

    // waves per GEMM = (M/64) * (N/32); 8 waves per block
    // 3) QKV GEMM (8192 x 768 x 256) with head scatter epilogue
    gemm_kernel<0><<<(ROWS / 64) * (3 * DMODEL / 32) / 8, 256, 0, stream>>>(
        x_h, win_h, in_proj_b, ROWS, 3 * DMODEL, DMODEL,
        nullptr, nullptr, nullptr, qh, kh, vt);

    // 4) fused biased attention + softmax + ctx
    attn_kernel<<<(BATCH * NHEAD) * (S_LEN / 16) / 8, 256, 0, stream>>>(
        qh, kh, vt, bias, ctx_h);

    // 5) out-proj (8192 x 256 x 256) + residual with x
    gemm_kernel<1><<<(ROWS / 64) * (DMODEL / 32) / 8, 256, 0, stream>>>(
        ctx_h, wout_h, out_b, ROWS, DMODEL, DMODEL,
        x2_f32, nullptr, x_f32, nullptr, nullptr, nullptr);

    // 6) LN2: y = LN(x2)
    ln_kernel<<<ROWS / 8, 256, 0, stream>>>(x2_f32, g2, be2, y_f32, y_h);

    // 7) FFN1 (8192 x 1024 x 256), ReLU -> f16
    gemm_kernel<2><<<(ROWS / 64) * (DFF / 32) / 8, 256, 0, stream>>>(
        y_h, w1_h, b1, ROWS, DFF, DMODEL,
        nullptr, h1_h, nullptr, nullptr, nullptr, nullptr);

    // 8) FFN2 (8192 x 256 x 1024) + residual with y -> d_out
    gemm_kernel<1><<<(ROWS / 64) * (DMODEL / 32) / 8, 256, 0, stream>>>(
        h1_h, w2_h, b2, ROWS, DMODEL, DFF,
        out, nullptr, y_f32, nullptr, nullptr, nullptr);
}
